// NPassReencModel_3204045603602
// MI455X (gfx1250) — compile-verified, matrix operand order checked by reading, not run on hardware
//
#include <hip/hip_runtime.h>
#include <hip/hip_bf16.h>

#define H 64
#define VOCAB 64
#define SLOTS 4
#define NPASS 3
#define B_ 128
#define L_ 4096
#define EPSF 1e-5f
#define SCALE 0.125f   // 1/sqrt(64)

typedef __attribute__((ext_vector_type(16))) __bf16 v16bf;
typedef __attribute__((ext_vector_type(2)))  __bf16 v2bf;
typedef __attribute__((ext_vector_type(16))) unsigned short v16us;
typedef __attribute__((ext_vector_type(8)))  unsigned v8u;
typedef __attribute__((ext_vector_type(8)))  float v8f;

#if defined(__has_builtin)
#if __has_builtin(__builtin_amdgcn_cvt_pk_bf16_f32)
#define HAVE_CVT_PK_BF16 1
#endif
#endif

__device__ __forceinline__ unsigned short f2bf(float f){
  unsigned u = __builtin_bit_cast(unsigned, f);
  unsigned r = u + 0x7FFFu + ((u >> 16) & 1u);   // round-to-nearest-even
  return (unsigned short)(r >> 16);
}

__device__ __forceinline__ unsigned pack2bf(float a, float b){
#ifdef HAVE_CVT_PK_BF16
  v2bf r = __builtin_amdgcn_cvt_pk_bf16_f32(a, b);   // v_cvt_pk_bf16_f32
  return __builtin_bit_cast(unsigned, r);
#else
  return (unsigned)f2bf(a) | ((unsigned)f2bf(b) << 16);
#endif
}

// B-matrix tile (32x16) pre-swizzled into lane-major layout: lane*16 contiguous bf16
__device__ __forceinline__ v16bf load_tile(const unsigned short* p, int lane){
  v16us u = *(const v16us*)(p + lane*16);
  return __builtin_bit_cast(v16bf, u);
}

// A-matrix 16x32 from row-major f32 in LDS. ISA layout: lane m=lane&15, h=lane>>4;
// element e: k = kBase + (e<8 ? 8h+e : 16+8h+(e-8)) -> two contiguous 8-float runs.
__device__ __forceinline__ v16bf load_A_lds(const float* base, int stride, int kBase, int lane){
  int m = lane & 15, h = lane >> 4;
  const float* p = base + m*stride + kBase + 8*h;
  v8u u;
  #pragma unroll
  for (int e = 0; e < 4; ++e) u[e]   = pack2bf(p[2*e],    p[2*e+1]);
  #pragma unroll
  for (int e = 0; e < 4; ++e) u[4+e] = pack2bf(p[16+2*e], p[16+2*e+1]);
  return __builtin_bit_cast(v16bf, u);
}

__device__ __forceinline__ v8f wmma_bf16(v16bf a, v16bf b, v8f c){
  return __builtin_amdgcn_wmma_f32_16x16x32_bf16(false, a, false, b, (short)0, c, false, false);
}

// ---------------- weight pre-swizzle into WMMA B-layout (bf16) ----------------
__global__ void prep_kernel(const float* ff1_w, const float* ff2_w, const float* g1_w,
                            unsigned short* ff1t, unsigned short* ff2t, unsigned short* g1t){
  int t = blockIdx.x*256 + threadIdx.x;
  if (t < 8192){                      // ff1: (64x128) -> kt<2, nt<8
    int T = t >> 9, r = t & 511, lane = r >> 4, e = r & 15;
    int kt = T >> 3, nt = T & 7;
    int h = lane >> 4, n = lane & 15;
    int k = kt*32 + ((e < 8) ? (8*h + e) : (16 + 8*h + (e-8)));
    ff1t[t] = f2bf(ff1_w[k*128 + nt*16 + n]);
  } else if (t < 16384){              // ff2: (128x64) -> kt<4, nt<4
    int tt = t - 8192;
    int T = tt >> 9, r = tt & 511, lane = r >> 4, e = r & 15;
    int kt = T >> 2, nt = T & 3;
    int h = lane >> 4, n = lane & 15;
    int k = kt*32 + ((e < 8) ? (8*h + e) : (16 + 8*h + (e-8)));
    ff2t[tt] = f2bf(ff2_w[k*64 + nt*16 + n]);
  } else if (t < 18432){              // g1: (64x32) -> kt<2, nt<2
    int tt = t - 16384;
    int T = tt >> 9, r = tt & 511, lane = r >> 4, e = r & 15;
    int kt = T >> 1, nt = T & 1;
    int h = lane >> 4, n = lane & 15;
    int k = kt*32 + ((e < 8) ? (8*h + e) : (16 + 8*h + (e-8)));
    g1t[tt] = f2bf(g1_w[k*32 + nt*16 + n]);
  }
}

// ---------------- encoder: embed + FFN(WMMA) + LN + gate(WMMA) ----------------
__global__ void __launch_bounds__(128) enc_kernel(
    const int* __restrict__ seq, const float* __restrict__ embed,
    const unsigned short* __restrict__ ff1t, const float* __restrict__ ff1_b,
    const unsigned short* __restrict__ ff2t, const float* __restrict__ ff2_b,
    const float* __restrict__ enc_g, const float* __restrict__ enc_b,
    const unsigned short* __restrict__ g1t, const float* __restrict__ g1_b,
    const float* __restrict__ g2_w, const float* __restrict__ g2_b,
    float* __restrict__ hidden, float* __restrict__ scores){
  __shared__ __align__(16) float sH[4][16][64];    // h, then res, then normalized
  __shared__ __align__(16) float sO1[4][16][128];  // ffn intermediate
  int w = threadIdx.x >> 5, lane = threadIdx.x & 31;
  int n0 = blockIdx.x*64 + w*16;                   // 16 tokens per wave

  // 1) gather embeddings -> sH; lane owns row (lane&15), half (lane>>4)
  {
    int m = lane & 15, hh = lane >> 4;
    int tok = seq[n0 + m];
    const float4* ep = (const float4*)(embed + tok*64 + hh*32);
    float4* sp = (float4*)&sH[w][m][hh*32];
    #pragma unroll
    for (int q=0;q<8;++q) sp[q] = ep[q];
  }
  __syncthreads();

  // 2) GEMM1: (16x64)@(64x128), K-tiled by 32
  v8f acc1[8];
  #pragma unroll
  for (int i=0;i<8;++i) acc1[i] = (v8f){0,0,0,0,0,0,0,0};
  #pragma unroll
  for (int kt=0; kt<2; ++kt){
    v16bf a = load_A_lds(&sH[w][0][0], 64, kt*32, lane);
    #pragma unroll
    for (int nt=0; nt<8; ++nt)
      acc1[nt] = wmma_bf16(a, load_tile(ff1t + (kt*8+nt)*512, lane), acc1[nt]);
  }
  {  // bias + relu -> sO1   (C layout: vgpr r => M = r + 8*(lane>>4), N = lane&15)
    int n = lane & 15, hh = lane >> 4;
    #pragma unroll
    for (int nt=0; nt<8; ++nt){
      float bias = ff1_b[nt*16+n];
      #pragma unroll
      for (int r=0;r<8;++r)
        sO1[w][r + hh*8][nt*16+n] = fmaxf(acc1[nt][r] + bias, 0.f);
    }
  }
  __syncthreads();

  // 3) GEMM2: (16x128)@(128x64)
  v8f acc2[4];
  #pragma unroll
  for (int i=0;i<4;++i) acc2[i] = (v8f){0,0,0,0,0,0,0,0};
  #pragma unroll
  for (int kt=0; kt<4; ++kt){
    v16bf a = load_A_lds(&sO1[w][0][0], 128, kt*32, lane);
    #pragma unroll
    for (int nt=0; nt<4; ++nt)
      acc2[nt] = wmma_bf16(a, load_tile(ff2t + (kt*4+nt)*512, lane), acc2[nt]);
  }
  {  // residual: sH <- h + ffn
    int n = lane & 15, hh = lane >> 4;
    #pragma unroll
    for (int nt=0; nt<4; ++nt){
      float bias = ff2_b[nt*16+n];
      #pragma unroll
      for (int r=0;r<8;++r)
        sH[w][r + hh*8][nt*16+n] += acc2[nt][r] + bias;
    }
  }
  __syncthreads();

  // 4) LayerNorm per row; lane handles row (lane&15), half (lane>>4) of the 64 cols
  {
    int m = lane & 15, hh = lane >> 4;
    float vbuf[32] __attribute__((aligned(16)));
    float s1 = 0.f, s2 = 0.f;
    #pragma unroll
    for (int c=0;c<32;++c){ float x = sH[w][m][hh*32+c]; vbuf[c]=x; s1+=x; s2+=x*x; }
    s1 += __shfl_xor(s1, 16); s2 += __shfl_xor(s2, 16);
    float mean = s1*(1.f/64.f);
    float var  = s2*(1.f/64.f) - mean*mean;
    float rstd = rsqrtf(var + EPSF);
    #pragma unroll
    for (int c=0;c<32;++c){
      int col = hh*32+c;
      float y = (vbuf[c]-mean)*rstd*enc_g[col] + enc_b[col];
      vbuf[c] = y; sH[w][m][col] = y;
    }
    float* gp = hidden + (size_t)(n0+m)*H + hh*32;
    #pragma unroll
    for (int q=0;q<8;++q) *(float4*)(gp + q*4) = *(float4*)&vbuf[q*4];
  }
  __syncthreads();

  // 5) gate MLP: relu(hidden@g1+b1)@g2+b2, sigmoid -> scores (g1 via WMMA)
  v8f gacc[2];
  #pragma unroll
  for (int i=0;i<2;++i) gacc[i] = (v8f){0,0,0,0,0,0,0,0};
  #pragma unroll
  for (int kt=0; kt<2; ++kt){
    v16bf a = load_A_lds(&sH[w][0][0], 64, kt*32, lane);
    #pragma unroll
    for (int nt=0; nt<2; ++nt)
      gacc[nt] = wmma_bf16(a, load_tile(g1t + (kt*2+nt)*512, lane), gacc[nt]);
  }
  {
    int n = lane & 15, hh = lane >> 4;
    float g2b = g2_b[0];
    float w0 = g2_w[n], w1 = g2_w[16+n];
    float b0 = g1_b[n], b1 = g1_b[16+n];
    #pragma unroll
    for (int r=0;r<8;++r){
      float s = fmaxf(gacc[0][r]+b0, 0.f)*w0 + fmaxf(gacc[1][r]+b1, 0.f)*w1;
      s += __shfl_xor(s,1); s += __shfl_xor(s,2);
      s += __shfl_xor(s,4); s += __shfl_xor(s,8);   // reduce 16-lane half
      if (n == r){
        int m = r + hh*8;
        scores[n0+m] = 1.f/(1.f + __expf(-(s + g2b)));
      }
    }
  }
}

// ---------------- top-4 per batch row, gather slots ----------------
__global__ void topk_kernel(const float* __restrict__ scores, const float* __restrict__ hidden,
                            int* __restrict__ idxOut, float* __restrict__ slots){
  int b = blockIdx.x, t = threadIdx.x;
  __shared__ float sv[1024];
  __shared__ int   si[1024];
  __shared__ int   fidx[SLOTS];
  float bv[4] = {-1e30f,-1e30f,-1e30f,-1e30f};
  int   bi[4] = {1<<30,1<<30,1<<30,1<<30};
  const float* srow = scores + (size_t)b*L_;
  for (int it=0; it<16; ++it){
    int l = t + it*256;
    float v = srow[l];
    if ((v > bv[3]) || (v == bv[3] && l < bi[3])){
      bv[3]=v; bi[3]=l;
      #pragma unroll
      for (int j=3;j>0;--j){
        bool sw = (bv[j] > bv[j-1]) || (bv[j]==bv[j-1] && bi[j] < bi[j-1]);
        if (sw){ float tv=bv[j]; bv[j]=bv[j-1]; bv[j-1]=tv;
                 int ti=bi[j]; bi[j]=bi[j-1]; bi[j-1]=ti; }
      }
    }
  }
  #pragma unroll
  for (int j=0;j<4;++j){ sv[t*4+j]=bv[j]; si[t*4+j]=bi[j]; }
  __syncthreads();
  for (int stride=128; stride>=1; stride>>=1){
    if (t < stride){
      float av[8]; int ai[8];
      #pragma unroll
      for (int j=0;j<4;++j){ av[j]=sv[t*4+j]; ai[j]=si[t*4+j];
                             av[4+j]=sv[(t+stride)*4+j]; ai[4+j]=si[(t+stride)*4+j]; }
      #pragma unroll
      for (int s=0;s<4;++s){
        int best=s;
        #pragma unroll
        for (int j=s+1;j<8;++j)
          if (av[j] > av[best] || (av[j]==av[best] && ai[j] < ai[best])) best=j;
        float tv=av[s]; av[s]=av[best]; av[best]=tv;
        int ti=ai[s]; ai[s]=ai[best]; ai[best]=ti;
      }
      #pragma unroll
      for (int j=0;j<4;++j){ sv[t*4+j]=av[j]; si[t*4+j]=ai[j]; }
    }
    __syncthreads();
  }
  if (t < SLOTS){ idxOut[b*SLOTS+t] = si[t]; fidx[t] = si[t]; }
  __syncthreads();
  int k = t >> 6, f = t & 63;
  slots[(b*SLOTS + k)*H + f] = hidden[((size_t)b*L_ + fidx[k])*H + f];
}

// -------- fold rk into slot queries: qt = rk_w @ (slots@rq_w+rq_b), c = Sq.rk_b --------
__global__ void slotproj_kernel(const float* __restrict__ slots,
                                const float* __restrict__ rq_w, const float* __restrict__ rq_b,
                                const float* __restrict__ rk_w, const float* __restrict__ rk_b,
                                float* __restrict__ qt, float* __restrict__ cvec, int p){
  int b = blockIdx.x, i = threadIdx.x;
  __shared__ float sSq[SLOTS][H];
  const float* rqw = rq_w + (size_t)p*H*H;
  const float* rkw = rk_w + (size_t)p*H*H;
  for (int k=0;k<SLOTS;++k){
    float v = rq_b[p*H + i];
    const float* sl = slots + (b*SLOTS + k)*H;
    for (int j=0;j<H;++j) v += sl[j]*rqw[j*H + i];
    sSq[k][i] = v;
  }
  __syncthreads();
  for (int k=0;k<SLOTS;++k){
    float v = 0.f;
    const float* row = rkw + i*H;
    for (int j=0;j<H;++j) v += row[j]*sSq[k][j];
    qt[(b*SLOTS + k)*H + i] = v * SCALE;
  }
  if (i < SLOTS){
    float c = 0.f;
    for (int j=0;j<H;++j) c += sSq[i][j]*rk_b[p*H + j];
    cvec[b*SLOTS + i] = c * SCALE;
  }
}

// ---------------- logits[b,k,l] = qt[b,k] . hidden[b,l] + c ----------------
__global__ void logits_kernel(const float* __restrict__ hidden, const float* __restrict__ qt,
                              const float* __restrict__ cvec, float* __restrict__ logits){
  int t = threadIdx.x;
  size_t n = (size_t)blockIdx.x*256 + t;       // global token; whole block same b
  int b = (int)(n >> 12);
  int l = (int)(n & 4095);
  __shared__ float sqt[SLOTS][H];
  __shared__ float sc[SLOTS];
  sqt[t>>6][t&63] = qt[(b*SLOTS + (t>>6))*H + (t&63)];
  if (t < SLOTS) sc[t] = cvec[b*SLOTS + t];
  __syncthreads();
  const float* hrow = hidden + n*H;
  float d0=sc[0], d1=sc[1], d2=sc[2], d3=sc[3];
  #pragma unroll 8
  for (int j=0;j<H;++j){
    float hv = hrow[j];
    d0 += hv*sqt[0][j]; d1 += hv*sqt[1][j];
    d2 += hv*sqt[2][j]; d3 += hv*sqt[3][j];
  }
  size_t base = ((size_t)b*SLOTS)*L_ + l;
  logits[base] = d0; logits[base + L_] = d1;
  logits[base + 2*(size_t)L_] = d2; logits[base + 3*(size_t)L_] = d3;
}

// -------- softmax over L + weighted sum of hidden for ALL 4 slots (one block / batch) --------
__global__ void attn_kernel(const float* __restrict__ logits, const float* __restrict__ hidden,
                            float* __restrict__ ctxn){
  int b = blockIdx.x, t = threadIdx.x;
  const float* lg = logits + (size_t)b*SLOTS*L_;
  __shared__ float red[256];
  __shared__ float mk[SLOTS], dk[SLOTS];
  __shared__ float wbuf[SLOTS][256];
  __shared__ __align__(16) float sacc[16][SLOTS][64];
  // per-slot max and exp-sum
  for (int k=0;k<SLOTS;++k){
    const float* row = lg + (size_t)k*L_;
    float m = -1e30f;
    for (int it=0; it<16; ++it) m = fmaxf(m, row[t + it*256]);
    red[t] = m; __syncthreads();
    for (int s=128; s>=1; s>>=1){ if (t < s) red[t] = fmaxf(red[t], red[t+s]); __syncthreads(); }
    if (t == 0) mk[k] = red[0];
    __syncthreads();
    float ps = 0.f;
    for (int it=0; it<16; ++it) ps += __expf(row[t + it*256] - mk[k]);
    red[t] = ps; __syncthreads();
    for (int s=128; s>=1; s>>=1){ if (t < s) red[t] += red[t+s]; __syncthreads(); }
    if (t == 0) dk[k] = red[0];
    __syncthreads();
  }
  // weighted accumulation, hidden[b] read exactly once; chunked weights in LDS
  int g = t >> 4, fi = t & 15;
  float4 a0 = {0,0,0,0}, a1 = {0,0,0,0}, a2 = {0,0,0,0}, a3 = {0,0,0,0};
  const float4* h4 = (const float4*)(hidden + (size_t)b*L_*H);
  float m0 = mk[0], m1 = mk[1], m2 = mk[2], m3 = mk[3];
  for (int chunk = 0; chunk < 16; ++chunk){
    int l0 = chunk*256;
    wbuf[0][t] = __expf(lg[l0 + t]           - m0);
    wbuf[1][t] = __expf(lg[L_ + l0 + t]      - m1);
    wbuf[2][t] = __expf(lg[2*L_ + l0 + t]    - m2);
    wbuf[3][t] = __expf(lg[3*L_ + l0 + t]    - m3);
    if (l0 + 256 < L_)
      __builtin_prefetch(h4 + (size_t)(l0 + 256 + t)*16, 0, 1);  // global_prefetch_b8
    __syncthreads();
    #pragma unroll
    for (int j=0;j<16;++j){
      int li = g + j*16;
      float4 hv = h4[(size_t)(l0 + li)*16 + fi];
      float w0 = wbuf[0][li], w1 = wbuf[1][li], w2 = wbuf[2][li], w3 = wbuf[3][li];
      a0.x += w0*hv.x; a0.y += w0*hv.y; a0.z += w0*hv.z; a0.w += w0*hv.w;
      a1.x += w1*hv.x; a1.y += w1*hv.y; a1.z += w1*hv.z; a1.w += w1*hv.w;
      a2.x += w2*hv.x; a2.y += w2*hv.y; a2.z += w2*hv.z; a2.w += w2*hv.w;
      a3.x += w3*hv.x; a3.y += w3*hv.y; a3.z += w3*hv.z; a3.w += w3*hv.w;
    }
    __syncthreads();
  }
  *(float4*)&sacc[g][0][fi*4] = a0;
  *(float4*)&sacc[g][1][fi*4] = a1;
  *(float4*)&sacc[g][2][fi*4] = a2;
  *(float4*)&sacc[g][3][fi*4] = a3;
  __syncthreads();
  {
    int k = t >> 6, f = t & 63;
    float s = 0.f;
    #pragma unroll
    for (int gg=0; gg<16; ++gg) s += sacc[gg][k][f];
    ctxn[(b*SLOTS + k)*H + f] = s/dk[k];
  }
}

// -------- slots = LN(slots + ctxn@rv_w + rv_b) --------
__global__ void update_kernel(const float* __restrict__ ctxn,
                              const float* __restrict__ rv_w, const float* __restrict__ rv_b,
                              const float* __restrict__ rn_g, const float* __restrict__ rn_b,
                              float* __restrict__ slots, int p){
  int b = blockIdx.x, i = threadIdx.x;
  __shared__ float tmp[SLOTS][H];
  __shared__ float stats[2*SLOTS];
  const float* rvw = rv_w + (size_t)p*H*H;
  for (int k=0;k<SLOTS;++k){
    float att = rv_b[p*H + i];
    const float* cx = ctxn + (b*SLOTS + k)*H;
    for (int j=0;j<H;++j) att += cx[j]*rvw[j*H + i];
    tmp[k][i] = slots[(b*SLOTS + k)*H + i] + att;
  }
  __syncthreads();
  if (i < SLOTS){
    float s1=0.f, s2=0.f;
    for (int j=0;j<H;++j){ float x = tmp[i][j]; s1+=x; s2+=x*x; }
    float mean = s1*(1.f/H); float var = s2*(1.f/H) - mean*mean;
    stats[2*i] = mean; stats[2*i+1] = rsqrtf(var + EPSF);
  }
  __syncthreads();
  for (int k=0;k<SLOTS;++k){
    float y = (tmp[k][i]-stats[2*k])*stats[2*k+1]*rn_g[p*H + i] + rn_b[p*H + i];
    slots[(b*SLOTS + k)*H + i] = y;
  }
}

// -------- read head + output projection --------
__global__ void read_kernel(const float* __restrict__ hidden, const float* __restrict__ slots,
                            const float* __restrict__ rd_q_w, const float* __restrict__ rd_q_b,
                            const float* __restrict__ out_w, const float* __restrict__ out_b,
                            float* __restrict__ out){
  int b = blockIdx.x, i = threadIdx.x;
  __shared__ float sq[H], sl[SLOTS], sread[H];
  const float* hlast = hidden + ((size_t)b*L_ + (L_-1))*H;
  float q = rd_q_b[i];
  for (int j=0;j<H;++j) q += hlast[j]*rd_q_w[j*H + i];
  sq[i] = q;
  __syncthreads();
  if (i < SLOTS){
    float d = 0.f;
    const float* s = slots + (b*SLOTS + i)*H;
    for (int j=0;j<H;++j) d += s[j]*sq[j];
    sl[i] = d * SCALE;
  }
  __syncthreads();
  if (i == 0){
    float m = fmaxf(fmaxf(sl[0],sl[1]), fmaxf(sl[2],sl[3]));
    float e0=__expf(sl[0]-m), e1=__expf(sl[1]-m), e2=__expf(sl[2]-m), e3=__expf(sl[3]-m);
    float inv = 1.f/(e0+e1+e2+e3);
    sl[0]=e0*inv; sl[1]=e1*inv; sl[2]=e2*inv; sl[3]=e3*inv;
  }
  __syncthreads();
  float r = 0.f;
  for (int k=0;k<SLOTS;++k) r += sl[k]*slots[(b*SLOTS+k)*H + i];
  sread[i] = r;
  __syncthreads();
  float o = out_b[i];
  for (int j=0;j<H;++j) o += sread[j]*out_w[j*VOCAB + i];
  out[b*VOCAB + i] = o;
}

extern "C" void kernel_launch(void* const* d_in, const int* in_sizes, int n_in,
                              void* d_out, int out_size, void* d_ws, size_t ws_size,
                              hipStream_t stream) {
  const int*   seq    = (const int*)  d_in[0];
  const float* embed  = (const float*)d_in[1];
  const float* ff1_w  = (const float*)d_in[2];
  const float* ff1_b  = (const float*)d_in[3];
  const float* ff2_w  = (const float*)d_in[4];
  const float* ff2_b  = (const float*)d_in[5];
  const float* enc_g  = (const float*)d_in[6];
  const float* enc_b  = (const float*)d_in[7];
  const float* g1_w   = (const float*)d_in[8];
  const float* g1_b   = (const float*)d_in[9];
  const float* g2_w   = (const float*)d_in[10];
  const float* g2_b   = (const float*)d_in[11];
  const float* rq_w   = (const float*)d_in[12];
  const float* rq_b   = (const float*)d_in[13];
  const float* rk_w   = (const float*)d_in[14];
  const float* rk_b   = (const float*)d_in[15];
  const float* rv_w   = (const float*)d_in[16];
  const float* rv_b   = (const float*)d_in[17];
  const float* rn_g   = (const float*)d_in[18];
  const float* rn_b   = (const float*)d_in[19];
  const float* rd_q_w = (const float*)d_in[20];
  const float* rd_q_b = (const float*)d_in[21];
  const float* out_w  = (const float*)d_in[22];
  const float* out_b  = (const float*)d_in[23];
  (void)in_sizes; (void)n_in; (void)out_size; (void)ws_size;
  float* out = (float*)d_out;

  char* ws = (char*)d_ws;
  size_t off = 0;
  auto alloc = [&](size_t bytes)->char*{
    char* p = ws + off; off = (off + bytes + 255) & ~(size_t)255; return p; };
  unsigned short* ff1t = (unsigned short*)alloc(8192*2);
  unsigned short* ff2t = (unsigned short*)alloc(8192*2);
  unsigned short* g1t  = (unsigned short*)alloc(2048*2);
  float* hidden = (float*)alloc((size_t)B_*L_*H*4);         // 128 MB (L2-resident)
  float* scores = (float*)alloc((size_t)B_*L_*4);
  int*   topidx = (int*)  alloc((size_t)B_*SLOTS*4);
  float* slots  = (float*)alloc((size_t)B_*SLOTS*H*4);
  float* qt     = (float*)alloc((size_t)B_*SLOTS*H*4);
  float* cvec   = (float*)alloc((size_t)B_*SLOTS*4);
  float* logits = (float*)alloc((size_t)B_*SLOTS*L_*4);     // 8 MB
  float* ctxn   = (float*)alloc((size_t)B_*SLOTS*H*4);

  prep_kernel<<<72, 256, 0, stream>>>(ff1_w, ff2_w, g1_w, ff1t, ff2t, g1t);
  enc_kernel<<<(B_*L_)/64, 128, 0, stream>>>(seq, embed, ff1t, ff1_b, ff2t, ff2_b,
                                             enc_g, enc_b, g1t, g1_b, g2_w, g2_b,
                                             hidden, scores);
  topk_kernel<<<B_, 256, 0, stream>>>(scores, hidden, topidx, slots);
  for (int p = 0; p < NPASS; ++p){
    slotproj_kernel<<<B_, 64, 0, stream>>>(slots, rq_w, rq_b, rk_w, rk_b, qt, cvec, p);
    logits_kernel<<<(B_*L_)/256, 256, 0, stream>>>(hidden, qt, cvec, logits);
    attn_kernel<<<B_, 256, 0, stream>>>(logits, hidden, ctxn);
    update_kernel<<<B_, 64, 0, stream>>>(ctxn, rv_w, rv_b, rn_g, rn_b, slots, p);
  }
  read_kernel<<<B_, 64, 0, stream>>>(hidden, slots, rd_q_w, rd_q_b, out_w, out_b, out);
}
